// SOMILanguageModel_33200097198712
// MI455X (gfx1250) — matrix operand order repaired and checked by reading, not compiled, not run on hardware
//
#include <hip/hip_runtime.h>
#include <hip/hip_bf16.h>
#include <stdint.h>

// ---------------- types / constants ----------------
typedef unsigned short u16_t;
typedef unsigned int   u32_t;
typedef __bf16 bf16_t;
typedef bf16_t v16bf __attribute__((ext_vector_type(16)));
typedef float  v8f   __attribute__((ext_vector_type(8)));

#define HDIM   256     // hidden size
#define BND    64      // bottleneck
#define TSEQ   2048    // sequence length
#define NLAYER 8
#define NSTEP  10
#define NT     16      // HDIM/16  (N tiles of force/gate GEMMs)
#define KTH    8       // HDIM/32  (K tiles, K=256)
#define HT     4       // BND/16   (N tiles of hidden GEMM)
#define KTB    2       // BND/32   (K tiles, K=64)
#define WAVES  4
#define TOK_PER_BLOCK 64
#define WELEMS (HDIM*HDIM + BND*HDIM + HDIM*BND)   // 98304 bf16 per layer

#define ALPHA1 1.0f
#define ALPHA0 0.1f
#define DTC    0.1f
#define BETAC  2.898275349237888f   // 2*sqrt(alpha1+alpha0+1)

union BF16x16 { v16bf v; u16_t u[16]; uint4 q[2]; };

__device__ __forceinline__ u16_t f2bf(float x) {
  return (u16_t)(__float_as_uint(x) >> 16);   // truncating f32->bf16
}
__device__ __forceinline__ float bf2f_lo(u32_t p) { return __uint_as_float(p << 16); }
__device__ __forceinline__ float bf2f_hi(u32_t p) { return __uint_as_float(p & 0xffff0000u); }

// B fragment (32x16, bf16): row-major [N][K] storage, K contiguous per lane.
// p = rowbase(N=lane&15) + kt*32 + half*16
__device__ __forceinline__ v16bf load_B16(const u16_t* p) {
  BF16x16 r;
  r.q[0] = *(const uint4*)(p);
  r.q[1] = *(const uint4*)(p + 8);
  return r.v;
}
// A fragment (16x32, bf16) from row-major [M][K] bf16 storage.
// p = rowbase(M=lane&15) + kt*32 + half*16 ; second 8 elems at +8
__device__ __forceinline__ v16bf load_A16(const u16_t* p) {
  BF16x16 r;
  r.q[0] = *(const uint4*)(p);
  r.q[1] = *(const uint4*)(p + 8);
  return r.v;
}

#define WMMA_BF16(A, B, C) \
  __builtin_amdgcn_wmma_f32_16x16x32_bf16(false, (A), false, (B), (short)0, (C), false, false)

// wave-private LDS RAW fence: wait DS ops + stop compiler reordering
#define LDS_FENCE() asm volatile("s_wait_dscnt 0" ::: "memory")

// ---------------- kernel 1a: embed table f32 -> bf16 (zero padded rows) ----
__global__ void somi_cvt_embed(const float* __restrict__ e, u16_t* __restrict__ ebf,
                               int n_valid, int n_total) {
  int i = blockIdx.x * 256 + threadIdx.x;
  if (i >= n_total) return;
  ebf[i] = (i < n_valid) ? f2bf(e[i]) : (u16_t)0;
}

// ---------------- kernel 1b: per-layer weight prep (once, not per block) ----
// Writes contiguous per-layer bf16 block: [ L_rw 256x256 | w1 64x256 | w2 256x64 ]
// L_rw = I - |W| / rowsum(|W|), row-major (matches WMMA B-operand layout).
__global__ __launch_bounds__(256)
void somi_prep_weights(const float* __restrict__ Wl,
                       const float* __restrict__ gw1,
                       const float* __restrict__ gw2,
                       u16_t* __restrict__ wbuf) {
  const int l = blockIdx.x;
  const int tid = threadIdx.x;
  u16_t* Ld  = wbuf + (size_t)l * WELEMS;
  u16_t* W1d = Ld  + HDIM * HDIM;
  u16_t* W2d = W1d + BND * HDIM;

  // one thread per Laplacian row
  {
    const int g = tid;
    const float4* row4 = (const float4*)(Wl + (size_t)l * HDIM * HDIM + (size_t)g * HDIM);
    float d = 0.f;
    for (int h4 = 0; h4 < HDIM / 4; ++h4) {
      float4 w = row4[h4];
      d += fabsf(w.x) + fabsf(w.y) + fabsf(w.z) + fabsf(w.w);
    }
    float inv = 1.0f / fmaxf(d, 1e-8f);
    uint2* dst = (uint2*)(Ld + g * HDIM);
    for (int h4 = 0; h4 < HDIM / 4; ++h4) {
      float4 w = row4[h4];
      int hbase = h4 * 4;
      u16_t e0 = f2bf(((g == hbase + 0) ? 1.0f : 0.0f) - fabsf(w.x) * inv);
      u16_t e1 = f2bf(((g == hbase + 1) ? 1.0f : 0.0f) - fabsf(w.y) * inv);
      u16_t e2 = f2bf(((g == hbase + 2) ? 1.0f : 0.0f) - fabsf(w.z) * inv);
      u16_t e3 = f2bf(((g == hbase + 3) ? 1.0f : 0.0f) - fabsf(w.w) * inv);
      dst[h4] = make_uint2((u32_t)e0 | ((u32_t)e1 << 16), (u32_t)e2 | ((u32_t)e3 << 16));
    }
  }
  // gate weights f32 -> bf16, same row-major layout as reference
  {
    const float4* w1 = (const float4*)(gw1 + (size_t)l * BND * HDIM);
    uint2* d1 = (uint2*)W1d;
    for (int i = tid; i < BND * HDIM / 4; i += 256) {
      float4 w = w1[i];
      d1[i] = make_uint2((u32_t)f2bf(w.x) | ((u32_t)f2bf(w.y) << 16),
                         (u32_t)f2bf(w.z) | ((u32_t)f2bf(w.w) << 16));
    }
    const float4* w2 = (const float4*)(gw2 + (size_t)l * HDIM * BND);
    uint2* d2 = (uint2*)W2d;
    for (int i = tid; i < HDIM * BND / 4; i += 256) {
      float4 w = w2[i];
      d2[i] = make_uint2((u32_t)f2bf(w.x) | ((u32_t)f2bf(w.y) << 16),
                         (u32_t)f2bf(w.z) | ((u32_t)f2bf(w.w) << 16));
    }
  }
}

// ---------------- kernel 2: full SOMI stack (embed -> 8 layers -> final LN) ----
// One wave owns a 16-token M-tile for the whole network.
// LDS: L_rw bf16 [256][256] + w1 bf16 [64][256] + w2 bf16 [256][64] (flat copy)
//      + per-wave scratch (bf16 [16][256] A-staging aliased with f32 [16][256]
//        layernorm staging) + per-wave hidden bf16 [16][64]
__global__ __launch_bounds__(128, 1)
void somi_settle(const int*   __restrict__ ids,
                 const float* __restrict__ embed,
                 const float* __restrict__ pos,
                 const u16_t* __restrict__ wbuf,
                 const float* __restrict__ masses,
                 const float* __restrict__ gb1,
                 const float* __restrict__ gb2,
                 const float* __restrict__ lng, const float* __restrict__ lnb,
                 const float* __restrict__ fng, const float* __restrict__ fnb,
                 u16_t* __restrict__ hb /* out: bf16 [B*T][H] */) {
  extern __shared__ unsigned char smem[];
  u16_t* Lb   = (u16_t*)smem;                         // 256*256 bf16
  u16_t* W1b  = Lb  + HDIM * HDIM;                    // 64*256
  u16_t* W2b  = W1b + BND * HDIM;                     // 256*64
  float* scAll= (float*)(W2b + HDIM * BND);           // WAVES*16*256 f32
  u16_t* hidAll=(u16_t*)(scAll + WAVES * 16 * HDIM);  // WAVES*16*64 bf16

  const int tid  = threadIdx.x;
  const int wave = tid >> 5;
  const int lane = tid & 31;
  const int half = lane >> 4;
  const int ln16 = lane & 15;

  float* sc  = scAll  + wave * 16 * HDIM;   // f32 view (layernorm staging)
  u16_t* scb = (u16_t*)sc;                  // bf16 view (per-step A staging)
  u16_t* hw  = hidAll + wave * 16 * BND;

  const int tok0 = blockIdx.x * TOK_PER_BLOCK + wave * 16; // global flat token base

  // integrator state, C/D layout: [nt][r] -> (M = r+half*8, N = nt*16+ln16)
  float phi[NT][8], phid[NT][8];
  u32_t tp[NT][4];   // target, bf16x2 packed (r=2i low, r=2i+1 high)

  // ---- init: token + position embedding directly into C layout ----
  #pragma unroll
  for (int nt = 0; nt < NT; ++nt) {
    int c = nt * 16 + ln16;
    #pragma unroll
    for (int i = 0; i < 4; ++i) {
      int t0 = tok0 + 2 * i + half * 8;
      float v0 = embed[(size_t)ids[t0]     * HDIM + c] + pos[(size_t)(t0       & (TSEQ-1)) * HDIM + c];
      float v1 = embed[(size_t)ids[t0 + 1] * HDIM + c] + pos[(size_t)((t0 + 1) & (TSEQ-1)) * HDIM + c];
      phi[nt][2*i]   = v0;
      phi[nt][2*i+1] = v1;
      tp[nt][i] = (u32_t)f2bf(v0) | ((u32_t)f2bf(v1) << 16);
    }
  }

  for (int l = 0; l < NLAYER; ++l) {
    __syncthreads();   // prior layer's consumers done before re-staging weights

    // ---- flat copy of the prepped bf16 weight block into LDS (192KB) ----
    {
      const uint4* src = (const uint4*)(wbuf + (size_t)l * WELEMS);
      uint4* dst = (uint4*)smem;
      for (int i = tid; i < WELEMS / 8; i += 128) dst[i] = src[i];
    }
    // prefetch next layer's weight block into cache while we compute
    if (l + 1 < NLAYER) {
      const char* nw = (const char*)(wbuf + (size_t)(l + 1) * WELEMS);
      for (int i = tid * 256; i < WELEMS * 2; i += 128 * 256)
        __builtin_prefetch(nw + i, 0, 2);
    }
    __syncthreads();

    // hoist per-column constants for this layer into registers
    float invm[NT], b2v[NT];
    #pragma unroll
    for (int nt = 0; nt < NT; ++nt) {
      int c = nt * 16 + ln16;
      invm[nt] = 1.0f / masses[l * HDIM + c];
      b2v[nt]  = gb2[l * HDIM + c];
    }
    float b1v[HT];
    #pragma unroll
    for (int ht = 0; ht < HT; ++ht) b1v[ht] = gb1[l * BND + ht * 16 + ln16];

    #pragma unroll
    for (int nt = 0; nt < NT; ++nt)
      #pragma unroll
      for (int r = 0; r < 8; ++r) phid[nt][r] = 0.f;

    // ---------------- settle: 10 semi-implicit Euler steps ----------------
    for (int s = 0; s < NSTEP; ++s) {
      // stage phi as bf16 to wave scratch, row-major [M][K]
      #pragma unroll
      for (int nt = 0; nt < NT; ++nt) {
        int c = nt * 16 + ln16;
        #pragma unroll
        for (int r = 0; r < 8; ++r) scb[(r + half * 8) * HDIM + c] = f2bf(phi[nt][r]);
      }
      LDS_FENCE();

      // cache all 8 A fragments of phi (reused by 16+4 N-tiles)
      v16bf Af[KTH];
      #pragma unroll
      for (int kt = 0; kt < KTH; ++kt)
        Af[kt] = load_A16(scb + ln16 * HDIM + kt * 32 + half * 16);

      // hidden = relu(phi @ w1^T + b1)  -> bf16 row-major [16][64]
      #pragma unroll
      for (int ht = 0; ht < HT; ++ht) {
        v8f acc = {};
        #pragma unroll
        for (int kt = 0; kt < KTH; ++kt) {
          v16bf B = load_B16(W1b + (ht * 16 + ln16) * HDIM + kt * 32 + half * 16);
          acc = WMMA_BF16(Af[kt], B, acc);
        }
        int ch = ht * 16 + ln16;
        #pragma unroll
        for (int r = 0; r < 8; ++r)
          hw[(r + half * 8) * BND + ch] = f2bf(fmaxf(acc[r] + b1v[ht], 0.f));
      }
      LDS_FENCE();

      v16bf Hf[KTB];
      #pragma unroll
      for (int kt = 0; kt < KTB; ++kt)
        Hf[kt] = load_A16(hw + ln16 * BND + kt * 32 + half * 16);

      // per output N-tile: gate, force, integrate
      #pragma unroll
      for (int nt = 0; nt < NT; ++nt) {
        int c = nt * 16 + ln16;
        v8f ga = {};
        #pragma unroll
        for (int kt = 0; kt < KTB; ++kt) {
          v16bf B = load_B16(W2b + c * BND + kt * 32 + half * 16);
          ga = WMMA_BF16(Hf[kt], B, ga);
        }
        v8f fa = {};
        #pragma unroll
        for (int kt = 0; kt < KTH; ++kt) {
          v16bf B = load_B16(Lb + c * HDIM + kt * 32 + half * 16);
          fa = WMMA_BF16(Af[kt], B, fa);
        }
        #pragma unroll
        for (int r = 0; r < 8; ++r) {
          float tv    = (r & 1) ? bf2f_hi(tp[nt][r >> 1]) : bf2f_lo(tp[nt][r >> 1]);
          float g     = 1.0f / (1.0f + __expf(-(ga[r] + b2v[nt])));
          float force = -ALPHA1 * fa[r] - ALPHA0 * phi[nt][r] + (tv - phi[nt][r]);
          float a     = g * force * invm[nt] - BETAC * phid[nt][r];
          phid[nt][r] += DTC * a;
          phi[nt][r]  += DTC * phid[nt][r];
        }
      }
    } // steps

    // ---- per-layer layernorm (per token, 16 lanes each own a token) ----
    #pragma unroll
    for (int nt = 0; nt < NT; ++nt) {
      int c = nt * 16 + ln16;
      #pragma unroll
      for (int r = 0; r < 8; ++r) sc[(r + half * 8) * HDIM + c] = phi[nt][r];
    }
    LDS_FENCE();
    {
      const float* g = lng + l * HDIM;
      const float* b = lnb + l * HDIM;
      if (lane < 16) {
        float* row = sc + lane * HDIM;
        float mu = 0.f;
        for (int c = 0; c < HDIM; ++c) mu += row[c];
        mu *= (1.0f / HDIM);
        float var = 0.f;
        for (int c = 0; c < HDIM; ++c) { float d = row[c] - mu; var += d * d; }
        float rs = rsqrtf(var * (1.0f / HDIM) + 1e-5f);
        for (int c = 0; c < HDIM; ++c) row[c] = (row[c] - mu) * rs * g[c] + b[c];
      }
    }
    LDS_FENCE();
    // reload: next layer's phi0 and target
    #pragma unroll
    for (int nt = 0; nt < NT; ++nt) {
      int c = nt * 16 + ln16;
      #pragma unroll
      for (int i = 0; i < 4; ++i) {
        float v0 = sc[(2*i     + half * 8) * HDIM + c];
        float v1 = sc[(2*i + 1 + half * 8) * HDIM + c];
        phi[nt][2*i]   = v0;
        phi[nt][2*i+1] = v1;
        tp[nt][i] = (u32_t)f2bf(v0) | ((u32_t)f2bf(v1) << 16);
      }
    }
  } // layers

  // ---- final layernorm (fn_g/fn_b), then emit bf16 activations ----
  if (lane < 16) {
    float* row = sc + lane * HDIM;
    float mu = 0.f;
    for (int c = 0; c < HDIM; ++c) mu += row[c];
    mu *= (1.0f / HDIM);
    float var = 0.f;
    for (int c = 0; c < HDIM; ++c) { float d = row[c] - mu; var += d * d; }
    float rs = rsqrtf(var * (1.0f / HDIM) + 1e-5f);
    for (int c = 0; c < HDIM; ++c) row[c] = (row[c] - mu) * rs * fng[c] + fnb[c];
  }
  LDS_FENCE();
  for (int m = 0; m < 16; ++m)
    for (int c = lane; c < HDIM; c += 32)
      hb[(size_t)(tok0 + m) * HDIM + c] = f2bf(sc[m * HDIM + c]);
}

// ---------------- kernel 3: logits = h @ embed^T  (bf16 WMMA, f32 out) ----
// Register blocking: each wave computes a 32x64 output block (2 M-tiles x
// 4 N-tiles = 8 WMMA tiles); per K-step 6 fragment loads feed 8 WMMAs.
// bf16 embed table (25.8MB) is fully L2-resident -> direct global operand loads.
#define LBM 2
#define LBN 4
__global__ __launch_bounds__(256)
void somi_logits(const u16_t* __restrict__ hb, const u16_t* __restrict__ ebf,
                 float* __restrict__ out, int V) {
  const int wave = threadIdx.x >> 5;
  const int lane = threadIdx.x & 31;
  const int half = lane >> 4;
  const int ln16 = lane & 15;
  const int mg = blockIdx.x * 8 + wave;   // 32-row group
  const int ng = blockIdx.y;              // 64-col group

  const u16_t* arow[LBM];
  const u16_t* brow[LBN];
  #pragma unroll
  for (int m = 0; m < LBM; ++m) arow[m] = hb  + (size_t)(mg * 32 + m * 16 + ln16) * HDIM;
  #pragma unroll
  for (int n = 0; n < LBN; ++n) brow[n] = ebf + (size_t)(ng * 64 + n * 16 + ln16) * HDIM;

  // pull operand rows toward the WGP while address setup finishes
  #pragma unroll
  for (int m = 0; m < LBM; ++m) __builtin_prefetch(arow[m], 0, 0);
  #pragma unroll
  for (int n = 0; n < LBN; ++n) __builtin_prefetch(brow[n], 0, 0);

  v8f acc[LBM][LBN] = {};
  #pragma unroll
  for (int kt = 0; kt < KTH; ++kt) {
    v16bf A[LBM], Bv[LBN];
    #pragma unroll
    for (int m = 0; m < LBM; ++m) A[m]  = load_A16(arow[m] + kt * 32 + half * 16);
    #pragma unroll
    for (int n = 0; n < LBN; ++n) Bv[n] = load_B16(brow[n] + kt * 32 + half * 16);
    #pragma unroll
    for (int m = 0; m < LBM; ++m)
      #pragma unroll
      for (int n = 0; n < LBN; ++n)
        acc[m][n] = WMMA_BF16(A[m], Bv[n], acc[m][n]);
  }

  #pragma unroll
  for (int n = 0; n < LBN; ++n) {
    int col = ng * 64 + n * 16 + ln16;
    if (col < V) {
      #pragma unroll
      for (int m = 0; m < LBM; ++m) {
        #pragma unroll
        for (int r = 0; r < 8; ++r) {
          int row = mg * 32 + m * 16 + r + half * 8;
          out[(size_t)row * V + col] = acc[m][n][r];
        }
      }
    }
  }
}

// ---------------- launch ----------------
extern "C" void kernel_launch(void* const* d_in, const int* in_sizes, int n_in,
                              void* d_out, int out_size, void* d_ws, size_t ws_size,
                              hipStream_t stream) {
  const int*   ids    = (const int*)  d_in[0];
  const float* embed  = (const float*)d_in[1];
  const float* pos    = (const float*)d_in[2];
  const float* Wl     = (const float*)d_in[3];
  const float* masses = (const float*)d_in[4];
  const float* gw1    = (const float*)d_in[5];
  const float* gb1    = (const float*)d_in[6];
  const float* gw2    = (const float*)d_in[7];
  const float* gb2    = (const float*)d_in[8];
  const float* lng    = (const float*)d_in[9];
  const float* lnb    = (const float*)d_in[10];
  const float* fng    = (const float*)d_in[11];
  const float* fnb    = (const float*)d_in[12];
  float* out = (float*)d_out;

  const int nTok  = in_sizes[0];            // B*T = 4096
  const int V     = in_sizes[1] / HDIM;     // 50257
  const int Ngrp  = (V + 63) / 64;          // 786 column groups
  const int Vpad  = Ngrp * 64;              // 50304

  u16_t* ebf  = (u16_t*)d_ws;                          // [Vpad][H] bf16
  u16_t* hbf  = ebf + (size_t)Vpad * HDIM;             // [nTok][H] bf16
  u16_t* wbuf = hbf + (size_t)nTok * HDIM;             // [8][WELEMS] bf16

  // 1a) embed table -> bf16 (padded with zeros)
  {
    int n_total = Vpad * HDIM, n_valid = V * HDIM;
    somi_cvt_embed<<<(n_total + 255) / 256, 256, 0, stream>>>(embed, ebf, n_valid, n_total);
  }
  // 1b) per-layer Laplacian + gate weights -> contiguous bf16 blocks (once)
  somi_prep_weights<<<NLAYER, 256, 0, stream>>>(Wl, gw1, gw2, wbuf);

  // 2) full SOMI stack; weights live in LDS (192KB) + per-wave scratch
  {
    size_t lds_bytes = (size_t)WELEMS * sizeof(u16_t)
                     + (size_t)WAVES * 16 * HDIM * sizeof(float)
                     + (size_t)WAVES * 16 * BND * sizeof(u16_t);   // 270,336 B < 320KB/WGP
    somi_settle<<<nTok / TOK_PER_BLOCK, 128, lds_bytes, stream>>>(
        ids, embed, pos, wbuf, masses, gb1, gb2, lng, lnb, fng, fnb, hbf);
  }

  // 3) tied lm_head GEMM: grid = (Mgroups/8 waves, Ngroups)
  {
    dim3 grid((nTok / 32) / 8, Ngrp);
    somi_logits<<<grid, 256, 0, stream>>>(hbf, ebf, out, V);
  }
}
